// MHAHead_35673998360865
// MI455X (gfx1250) — compile-verified, matrix operand order checked by reading, not run on hardware
//
#include <hip/hip_runtime.h>
#include <cstdint>
#include <cstddef>

// ---------- types for CDNA5 WMMA (wave32, 16x16x32 bf16 -> f32) ----------
typedef __bf16 bf16_t;
typedef __bf16 v16bf __attribute__((ext_vector_type(16)));
typedef __bf16 v8bf  __attribute__((ext_vector_type(8)));
typedef float  v8f   __attribute__((ext_vector_type(8)));

#define ZV8F {0.f,0.f,0.f,0.f,0.f,0.f,0.f,0.f}

__device__ __forceinline__ v16bf frag16(const bf16_t* p0, const bf16_t* p1) {
    v8bf lo = *(const v8bf*)p0;   // 128-bit load (K = g*8 .. g*8+7)
    v8bf hi = *(const v8bf*)p1;   // 128-bit load (K = 16+g*8 .. 16+g*8+7)
    return __builtin_shufflevector(lo, hi, 0,1,2,3,4,5,6,7,8,9,10,11,12,13,14,15);
}

__device__ __forceinline__ v8f wmma_bf16(v16bf a, v16bf b, v8f c) {
    return __builtin_amdgcn_wmma_f32_16x16x32_bf16(false, a, false, b, (short)0, c,
                                                   false, false);
}

// ---------- problem constants ----------
static constexpr int Dv  = 768;
static constexpr int D   = 512;
static constexpr int Bv  = 32;
static constexpr int Lv  = 196;
static constexpr int LVP = 224;     // 14*16, divisible by 32
static constexpr int Bt  = 32;
static constexpr int Lt  = 64;
static constexpr int MV  = Bv * Lv;   // 6272
static constexpr int MVP = Bv * LVP;  // 7168
static constexpr int MT  = Bt * Lt;   // 2048

// ---------- f32 -> bf16 conversion ----------
__global__ void k_f32_to_bf16(const float* __restrict__ in, bf16_t* __restrict__ out, int n) {
    int i = blockIdx.x * blockDim.x + threadIdx.x;
    int stride = gridDim.x * blockDim.x;
    for (; i < n; i += stride) out[i] = (bf16_t)in[i];
}

// ---------- GEMM: C[M,N] = A[M,K](bf16,rowmaj) x W[N,K](bf16,rowmaj)^T + bias ----------
// one wave computes a 16x32 output tile (two 16x16 WMMA tiles, A fragment reused)
__global__ void k_gemm(const bf16_t* __restrict__ A, const bf16_t* __restrict__ W,
                       const float* __restrict__ bias, float* __restrict__ C,
                       int M, int N, int K) {
    int lane = threadIdx.x & 31;
    int wid  = blockIdx.x * (blockDim.x >> 5) + (threadIdx.x >> 5);
    int tilesN2 = N >> 5;
    int mt = wid / tilesN2;
    int nt = wid % tilesN2;
    int m0 = mt * 16;
    if (m0 >= M) return;
    int n0 = nt * 32;
    int r = lane & 15, g = lane >> 4;

    const bf16_t* ar  = A + (size_t)(m0 + r) * K;
    const bf16_t* br0 = W + (size_t)(n0 + r) * K;
    const bf16_t* br1 = W + (size_t)(n0 + 16 + r) * K;

    v8f a0 = ZV8F, a1 = ZV8F;
    for (int kk = 0; kk < K; kk += 32) {
        v16bf af = frag16(ar  + kk + g*8, ar  + kk + 16 + g*8);
        v16bf b0 = frag16(br0 + kk + g*8, br0 + kk + 16 + g*8);
        v16bf b1 = frag16(br1 + kk + g*8, br1 + kk + 16 + g*8);
        a0 = wmma_bf16(af, b0, a0);
        a1 = wmma_bf16(af, b1, a1);
    }
    int c0 = n0 + r, c1 = n0 + 16 + r, rb = g * 8;
    float bv0 = bias ? bias[c0] : 0.f;
    float bv1 = bias ? bias[c1] : 0.f;
    #pragma unroll
    for (int j = 0; j < 8; ++j) {
        C[(size_t)(m0 + rb + j) * N + c0] = a0[j] + bv0;
        C[(size_t)(m0 + rb + j) * N + c1] = a1[j] + bv1;
    }
}

// ---------- row L2-norm (+ optional transposed raw-value output for V matrices) ----------
// Input rows addressed as b*Lin + l (l < Lreal); outputs padded to Lpad rows.
// outN[b*Lpad+l][:] = bf16(x / max(||x||,eps));  outT[b][d][l] = bf16(x[d]) (raw).
__global__ void k_norm_rows(const float* __restrict__ X, bf16_t* __restrict__ outN,
                            bf16_t* __restrict__ outT, int Lreal, int Lpad, int Lin) {
    int rb = blockIdx.x;                 // b*Lpad + l
    int b = rb / Lpad, l = rb % Lpad;
    int tid = threadIdx.x;
    __shared__ float wsum[8];
    if (l >= Lreal) {
        bf16_t z = (bf16_t)0.f;
        outN[(size_t)rb * D + tid]       = z;
        outN[(size_t)rb * D + tid + 256] = z;
        if (outT) {
            outT[((size_t)b * D + tid)       * Lpad + l] = z;
            outT[((size_t)b * D + tid + 256) * Lpad + l] = z;
        }
        return;
    }
    const float* x = X + ((size_t)b * Lin + l) * D;
    float x0 = x[tid], x1 = x[tid + 256];
    float ss = x0 * x0 + x1 * x1;
    #pragma unroll
    for (int m = 1; m < 32; m <<= 1) ss += __shfl_xor(ss, m, 32);
    if ((tid & 31) == 0) wsum[tid >> 5] = ss;
    __syncthreads();
    float tot = wsum[0] + wsum[1] + wsum[2] + wsum[3] + wsum[4] + wsum[5] + wsum[6] + wsum[7];
    float inv = 1.f / fmaxf(sqrtf(tot), 1e-12f);
    outN[(size_t)rb * D + tid]       = (bf16_t)(x0 * inv);
    outN[(size_t)rb * D + tid + 256] = (bf16_t)(x1 * inv);
    if (outT) {
        outT[((size_t)b * D + tid)       * Lpad + l] = (bf16_t)x0;
        outT[((size_t)b * D + tid + 256) * Lpad + l] = (bf16_t)x1;
    }
}

// ---------- fused cross-attention + row-l2norm + row-sum reduction ----------
// grid = (32 q-batches, 32 k-batches); 256 threads = 8 waves per block.
// Per (qb,kb) pair, in chunks of 16 query rows:
//   S = relu(q.kT)*20 (+ key mask / pad mask -> -1e9), softmax over keys -> P(bf16,LDS)
//   O = P x V   (V supplied transposed as VT[512][Lk_pad])
//   rowss = ||O_row||^2 via shfl reduce + LDS atomics; rnorm = keep/||O_row||
//   accum[d] += sum_rows O[row][d]*rnorm[row]   (the collapsed mean-einsum)
// Out[(kb*32+qb)][0..511] = accum (divide by 196*64 in the final dot kernel).
__global__ void k_attn(const bf16_t* __restrict__ Q, const bf16_t* __restrict__ Kd,
                       const bf16_t* __restrict__ VT,
                       const int* __restrict__ kmask_base, const int* __restrict__ rmask_base,
                       int Lq_real, int Lq_pad, int Lk_real, int Lk_pad,
                       float* __restrict__ Out) {
    const int qb = blockIdx.x, kb = blockIdx.y;
    const bf16_t* q  = Q  + (size_t)qb * Lq_pad * D;
    const bf16_t* k  = Kd + (size_t)kb * Lk_pad * D;
    const bf16_t* vT = VT + (size_t)kb * (size_t)D * Lk_pad;
    const int* kmask = kmask_base ? kmask_base + (size_t)kb * Lk_real : nullptr;
    const int* rmask = rmask_base ? rmask_base + (size_t)qb * Lq_real : nullptr;

    __shared__ __align__(16) float  sS[16][LVP];   // raw scores for one 16-row chunk
    __shared__ __align__(16) bf16_t sP[16][LVP];   // softmax probs (bf16 for WMMA A)
    __shared__ float rowss[16];
    __shared__ float rnorm[16];
    __shared__ float accum[D];

    const int tid  = threadIdx.x;
    const int lane = tid & 31;
    const int wid  = tid >> 5;
    const int r = lane & 15, g = lane >> 4;
    const int nKt = Lk_pad >> 4;       // key tiles (<=14)

    accum[tid] = 0.f; accum[tid + 256] = 0.f;
    __syncthreads();

    for (int r0 = 0; r0 < Lq_pad; r0 += 16) {
        if (r0 >= Lq_real) break;      // fully-padded chunks contribute nothing

        // ---- 1) S = q x k^T for this 16-row chunk ----
        const bf16_t* qrow = q + (size_t)(r0 + r) * D;
        for (int nt = wid; nt < nKt; nt += 8) {
            const bf16_t* krow = k + (size_t)(nt * 16 + r) * D;
            v8f acc = ZV8F;
            for (int kk = 0; kk < D; kk += 32) {
                v16bf af = frag16(qrow + kk + g*8, qrow + kk + 16 + g*8);
                v16bf bf = frag16(krow + kk + g*8, krow + kk + 16 + g*8);
                acc = wmma_bf16(af, bf, acc);
            }
            int col = nt * 16 + r, rbase = g * 8;
            #pragma unroll
            for (int j = 0; j < 8; ++j) sS[rbase + j][col] = acc[j];
        }
        __syncthreads();
        if (tid < 16) rowss[tid] = 0.f;

        // ---- 2) relu*20 + mask + softmax over keys; write bf16 P ----
        {
            int row = tid >> 4, sub = tid & 15;
            int cnt = Lk_pad >> 4;     // <= 14
            float vv[14];
            float mx = -3.0e38f;
            for (int i = 0; i < cnt; ++i) {
                int c = sub + i * 16;
                float s = fmaxf(sS[row][c], 0.f) * 20.f;
                bool dead = (c >= Lk_real);
                if (!dead && kmask) dead = (kmask[c] != 0);
                if (dead) s = -1.0e9f;
                vv[i] = s;
                mx = fmaxf(mx, s);
            }
            #pragma unroll
            for (int m = 1; m < 16; m <<= 1) mx = fmaxf(mx, __shfl_xor(mx, m, 32));
            float sum = 0.f;
            for (int i = 0; i < cnt; ++i) { float e = __expf(vv[i] - mx); vv[i] = e; sum += e; }
            #pragma unroll
            for (int m = 1; m < 16; m <<= 1) sum += __shfl_xor(sum, m, 32);
            float inv = 1.f / sum;
            for (int i = 0; i < cnt; ++i) sP[row][sub + i * 16] = (bf16_t)(vv[i] * inv);
        }
        __syncthreads();

        // ---- 3) O = P x V ; per-row sum-of-squares ----
        v8f oacc[4];
        #pragma unroll
        for (int i = 0; i < 4; ++i) {
            int nt = wid * 4 + i;      // 32 d-tiles, 4 per wave
            const bf16_t* vrow = vT + (size_t)(nt * 16 + r) * Lk_pad;
            v8f acc = ZV8F;
            for (int kk = 0; kk < Lk_pad; kk += 32) {
                v16bf af = frag16(&sP[r][kk + g*8], &sP[r][kk + 16 + g*8]);
                v16bf bf = frag16(vrow + kk + g*8, vrow + kk + 16 + g*8);
                acc = wmma_bf16(af, bf, acc);
            }
            oacc[i] = acc;
            #pragma unroll
            for (int j = 0; j < 8; ++j) {
                float s = acc[j] * acc[j];
                #pragma unroll
                for (int m = 1; m < 16; m <<= 1) s += __shfl_xor(s, m, 32);
                if ((lane & 15) == 0) atomicAdd(&rowss[g * 8 + j], s);
            }
        }
        __syncthreads();

        // ---- 4) per-row 1/norm with row masking ----
        if (tid < 16) {
            int grow = r0 + tid;
            float inv = 1.f / fmaxf(sqrtf(rowss[tid]), 1e-12f);
            bool keep = (grow < Lq_real);
            if (keep && rmask) keep = (rmask[grow] == 0);
            rnorm[tid] = keep ? inv : 0.f;
        }
        __syncthreads();

        // ---- 5) accum[d] += sum over rows of normalized O ----
        #pragma unroll
        for (int i = 0; i < 4; ++i) {
            int d = (wid * 4 + i) * 16 + r;
            float s = 0.f;
            #pragma unroll
            for (int j = 0; j < 8; ++j) s += oacc[i][j] * rnorm[g * 8 + j];
            atomicAdd(&accum[d], s);
        }
        __syncthreads();
    }

    float* o = Out + ((size_t)kb * gridDim.x + qb) * D;
    o[tid] = accum[tid];
    o[tid + 256] = accum[tid + 256];
}

// ---------- final: out[t*32+v] = dot(Asum[t,v], Bsum[v,t]) / (196*64) ----------
__global__ void k_final(const float* __restrict__ As, const float* __restrict__ Bs,
                        float* __restrict__ out) {
    int p = blockIdx.x;
    int t = p >> 5, v = p & 31;
    int tid = threadIdx.x;
    const float* a = As + (size_t)p * D;
    const float* b = Bs + (size_t)(v * 32 + t) * D;
    float s = a[tid] * b[tid] + a[tid + 256] * b[tid + 256];
    #pragma unroll
    for (int m = 1; m < 32; m <<= 1) s += __shfl_xor(s, m, 32);
    __shared__ float wsum[8];
    if ((tid & 31) == 0) wsum[tid >> 5] = s;
    __syncthreads();
    if (tid == 0) {
        float tot = 0.f;
        for (int i = 0; i < 8; ++i) tot += wsum[i];
        out[p] = tot * (1.0f / (196.0f * 64.0f));
    }
}

// ---------- host orchestration ----------
extern "C" void kernel_launch(void* const* d_in, const int* in_sizes, int n_in,
                              void* d_out, int out_size, void* d_ws, size_t ws_size,
                              hipStream_t stream) {
    (void)in_sizes; (void)n_in; (void)out_size; (void)ws_size;

    const float* vis   = (const float*)d_in[0];
    const float* txt   = (const float*)d_in[1];
    const int*   mask  = (const int*)  d_in[2];
    const float* W_vis = (const float*)d_in[3];
    const float* b_vis = (const float*)d_in[4];
    const float* W_txt = (const float*)d_in[5];
    const float* b_txt = (const float*)d_in[6];
    const float* W_qs  = (const float*)d_in[7];
    const float* W_ks  = (const float*)d_in[8];
    float* out = (float*)d_out;

    char* ws = (char*)d_ws;
    size_t off = 0;
    auto carve = [&](size_t bytes) -> char* {
        char* p = ws + off;
        off = (off + bytes + 255) & ~(size_t)255;
        return p;
    };

    bf16_t* visbf  = (bf16_t*)carve((size_t)MV  * Dv * 2);
    bf16_t* txtbf  = (bf16_t*)carve((size_t)MT  * Dv * 2);
    bf16_t* Wvisbf = (bf16_t*)carve((size_t)D * Dv * 2);
    bf16_t* Wtxtbf = (bf16_t*)carve((size_t)D * Dv * 2);
    bf16_t* Wqsbf  = (bf16_t*)carve((size_t)D * D  * 2);
    bf16_t* Wksbf  = (bf16_t*)carve((size_t)D * D  * 2);
    float*  scr    = (float*) carve((size_t)MVP * D * 4);   // reused f32 GEMM output
    bf16_t* vnorm  = (bf16_t*)carve((size_t)MVP * D * 2);
    bf16_t* tnorm  = (bf16_t*)carve((size_t)MT  * D * 2);
    bf16_t* vvT    = (bf16_t*)carve((size_t)Bv * D * LVP * 2);
    bf16_t* vtT    = (bf16_t*)carve((size_t)Bt * D * Lt  * 2);
    bf16_t* qv     = (bf16_t*)carve((size_t)MVP * D * 2);
    bf16_t* kv     = (bf16_t*)carve((size_t)MVP * D * 2);
    bf16_t* qt     = (bf16_t*)carve((size_t)MT  * D * 2);
    bf16_t* kt     = (bf16_t*)carve((size_t)MT  * D * 2);
    float*  Asum   = (float*) carve((size_t)1024 * D * 4);
    float*  Bsum   = (float*) carve((size_t)1024 * D * 4);

    auto conv = [&](const float* src, bf16_t* dst, int n) {
        int blocks = (n + 255) / 256; if (blocks > 4096) blocks = 4096;
        k_f32_to_bf16<<<blocks, 256, 0, stream>>>(src, dst, n);
    };
    auto gemm = [&](const bf16_t* A, const bf16_t* W, const float* bias, float* C,
                    int M, int N, int K) {
        int tiles = (M / 16) * (N / 32);
        k_gemm<<<(tiles + 7) / 8, 256, 0, stream>>>(A, W, bias, C, M, N, K);
    };

    // 1) bf16 staging
    conv(vis,   visbf,  MV * Dv);
    conv(txt,   txtbf,  MT * Dv);
    conv(W_vis, Wvisbf, D * Dv);
    conv(W_txt, Wtxtbf, D * Dv);
    conv(W_qs,  Wqsbf,  D * D);
    conv(W_ks,  Wksbf,  D * D);

    // 2) embeddings + norms (+ transposed raw V)
    gemm(visbf, Wvisbf, b_vis, scr, MV, D, Dv);
    k_norm_rows<<<MVP, 256, 0, stream>>>(scr, vnorm, vvT, Lv, LVP, Lv);
    gemm(txtbf, Wtxtbf, b_txt, scr, MT, D, Dv);
    k_norm_rows<<<MT, 256, 0, stream>>>(scr, tnorm, vtT, Lt, Lt, Lt);

    // 3) q/k projections + norms (padded rows are zero -> stay zero)
    gemm(vnorm, Wqsbf, nullptr, scr, MVP, D, D);
    k_norm_rows<<<MVP, 256, 0, stream>>>(scr, qv, nullptr, LVP, LVP, LVP);
    gemm(vnorm, Wksbf, nullptr, scr, MVP, D, D);
    k_norm_rows<<<MVP, 256, 0, stream>>>(scr, kv, nullptr, LVP, LVP, LVP);
    gemm(tnorm, Wqsbf, nullptr, scr, MT, D, D);
    k_norm_rows<<<MT, 256, 0, stream>>>(scr, qt, nullptr, Lt, Lt, Lt);
    gemm(tnorm, Wksbf, nullptr, scr, MT, D, D);
    k_norm_rows<<<MT, 256, 0, stream>>>(scr, kt, nullptr, Lt, Lt, Lt);

    // 4) fused attention (both directions), reduced to per-pair 512-vectors
    dim3 agrid(32, 32);
    // att_text: q=visual, k/v=textual, key mask from text padding -> Asum[t*32+v]
    k_attn<<<agrid, 256, 0, stream>>>(qv, kt, vtT, mask, nullptr,
                                      Lv, LVP, Lt, Lt, Asum);
    // att_vis: q=textual, k/v=visual, row mask zeroes masked text rows -> Bsum[v*32+t]
    k_attn<<<agrid, 256, 0, stream>>>(qt, kv, vvT, nullptr, mask,
                                      Lt, Lt, Lv, LVP, Bsum);

    // 5) final dot + global mean scale
    k_final<<<1024, 256, 0, stream>>>(Asum, Bsum, out);
}
